// MoELayer_16277926052453
// MI455X (gfx1250) — matrix operand order repaired and checked by reading, not compiled, not run on hardware
//
#include <hip/hip_runtime.h>
#include <hip/hip_bf16.h>

// Problem dims (from reference): B=2,S=2048 -> T=4096 tokens, D=768, E=8, F=3072
#define T_TOK 4096
#define D_DIM 768
#define E_NUM 8
#define F_DIM 3072
#define TILE_M 32
#define FC 64                 // F-chunk for fused h1 -> out
#define W1S_ROW (D_DIM + 8)   // 776 elems -> 1552B rows, 16B aligned
#define W2S_ROW (FC + 8)      // 72 elems  -> 144B rows, 16B aligned

typedef __attribute__((ext_vector_type(8)))  __bf16 v8bf;
typedef __attribute__((ext_vector_type(16))) __bf16 v16bf;
typedef __attribute__((ext_vector_type(8)))  float  v8f;

// ---- WMMA operand loaders (bf16, 16x16x32) -------------------------------
// A matrix 16x32 bf16: lanes 0-15 = rows M, VGPR0-3 K=0..7, VGPR4-7 K=16..23;
// lanes 16-31 same M, K=8..15 and K=24..31.
__device__ __forceinline__ v16bf lds_load_a(const __hip_bfloat16* base, int row_elems,
                                            int m0, int ks, int lane) {
  const int half = lane >> 4;                  // 0 or 1
  const int m = m0 + (lane & 15);
  const __hip_bfloat16* p = base + (size_t)m * row_elems + ks + half * 8;
  v8bf lo = *reinterpret_cast<const v8bf*>(p);        // K = ks + half*8 .. +7
  v8bf hi = *reinterpret_cast<const v8bf*>(p + 16);   // K = ks + 16 + half*8 .. +7
  return __builtin_shufflevector(lo, hi, 0,1,2,3,4,5,6,7,8,9,10,11,12,13,14,15);
}

// B matrix 32x16 bf16 (stored transposed in LDS as [n][k]):
// lanes 0-15: column n, K=0..7 and K=16..23; lanes 16-31: K=8..15 and K=24..31.
__device__ __forceinline__ v16bf lds_load_b(const __hip_bfloat16* base, int row_elems,
                                            int n0, int ks, int lane) {
  const int n = n0 + (lane & 15);
  const __hip_bfloat16* p = base + (size_t)n * row_elems + ks + (lane >> 4) * 8;
  v8bf lo = *reinterpret_cast<const v8bf*>(p);
  v8bf hi = *reinterpret_cast<const v8bf*>(p + 16);
  return __builtin_shufflevector(lo, hi, 0,1,2,3,4,5,6,7,8,9,10,11,12,13,14,15);
}

// ---- Kernel 0: zero per-expert token counters ----------------------------
__global__ void moe_zero_counts(int* counts) {
  if (threadIdx.x < E_NUM) counts[threadIdx.x] = 0;
}

// ---- Kernel 1: router (logits, softmax-max prob, argmax, bucketing) ------
// One wave32 per token; lanes stride over D.
__global__ void moe_router(const float* __restrict__ x,
                           const float* __restrict__ w_router,
                           float* __restrict__ logits_out,
                           int* __restrict__ idx_out,
                           float* __restrict__ prob_ws,
                           int* __restrict__ counts,
                           int* __restrict__ bucket) {
  const int wave = threadIdx.x >> 5;
  const int lane = threadIdx.x & 31;
  const int tok = blockIdx.x * 8 + wave;
  if (tok >= T_TOK) return;

  float acc[E_NUM];
#pragma unroll
  for (int e = 0; e < E_NUM; ++e) acc[e] = 0.f;

  const float* xr = x + (size_t)tok * D_DIM;
  for (int d = lane; d < D_DIM; d += 32) {
    const float xv = xr[d];
    const float4* w4 = reinterpret_cast<const float4*>(w_router + (size_t)d * E_NUM);
    float4 a = w4[0], b = w4[1];
    acc[0] += xv * a.x; acc[1] += xv * a.y; acc[2] += xv * a.z; acc[3] += xv * a.w;
    acc[4] += xv * b.x; acc[5] += xv * b.y; acc[6] += xv * b.z; acc[7] += xv * b.w;
  }
#pragma unroll
  for (int off = 16; off >= 1; off >>= 1)
#pragma unroll
    for (int e = 0; e < E_NUM; ++e) acc[e] += __shfl_xor(acc[e], off, 32);

  if (lane < E_NUM) logits_out[(size_t)tok * E_NUM + lane] = acc[lane];

  if (lane == 0) {
    float mx = acc[0]; int am = 0;
#pragma unroll
    for (int e = 1; e < E_NUM; ++e)
      if (acc[e] > mx) { mx = acc[e]; am = e; }
    float s = 0.f;
#pragma unroll
    for (int e = 0; e < E_NUM; ++e) s += __expf(acc[e] - mx);
    prob_ws[tok] = 1.0f / s;        // == max softmax prob
    idx_out[tok] = am;
    int pos = atomicAdd(&counts[am], 1);
    bucket[am * T_TOK + pos] = tok;
  }
}

// ---- Kernel 2: grouped expert FFN, bf16 WMMA, fused relu(X@W1)@W2 --------
__global__ __launch_bounds__(256)
void moe_expert_ffn(const float* __restrict__ x,
                    const float* __restrict__ w1,
                    const float* __restrict__ w2,
                    const float* __restrict__ prob_ws,
                    const int* __restrict__ counts,
                    const int* __restrict__ bucket,
                    float* __restrict__ out) {
  const int e = blockIdx.y;
  const int count = counts[e];
  const int base = blockIdx.x * TILE_M;
  if (base >= count) return;
  const int rows = min(TILE_M, count - base);

  __shared__ __align__(16) __hip_bfloat16 Xs[TILE_M * D_DIM];    // 48 KB
  __shared__ __align__(16) __hip_bfloat16 W1s[FC * W1S_ROW];     // ~97 KB  [n][k]
  __shared__ __align__(16) __hip_bfloat16 W2s[D_DIM * W2S_ROW];  // ~108 KB [n][k]
  __shared__ __align__(16) __hip_bfloat16 H1s[TILE_M * FC];      // 4 KB
  __shared__ int   tok_s[TILE_M];
  __shared__ float prob_s[TILE_M];

  const int tid = threadIdx.x;
  const int wv = tid >> 5;
  const int lane = tid & 31;

  if (tid < TILE_M) {
    int t = (tid < rows) ? bucket[e * T_TOK + base + tid] : -1;
    tok_s[tid] = t;
    prob_s[tid] = (t >= 0) ? prob_ws[t] : 0.f;
  }
  __syncthreads();

  // Stage X tile (fp32 -> bf16, float4 reads, packed bf16x2 stores).
  for (int i = tid * 4; i < TILE_M * D_DIM; i += 256 * 4) {
    const int r = i / D_DIM, d = i - r * D_DIM;
    float4 v = make_float4(0.f, 0.f, 0.f, 0.f);
    if (r < rows)
      v = *reinterpret_cast<const float4*>(&x[(size_t)tok_s[r] * D_DIM + d]);
    __hip_bfloat162 p0 = __float22bfloat162_rn(make_float2(v.x, v.y));
    __hip_bfloat162 p1 = __float22bfloat162_rn(make_float2(v.z, v.w));
    *reinterpret_cast<__hip_bfloat162*>(&Xs[i])     = p0;
    *reinterpret_cast<__hip_bfloat162*>(&Xs[i + 2]) = p1;
  }

  // Persistent output accumulators: wave owns rowtile rt, 12 column tiles.
  const int rt = wv & 1;
  const int cbase = (wv >> 1) * 12;
  v8f acc[12];
#pragma unroll
  for (int t = 0; t < 12; ++t) acc[t] = (v8f)(0.f);

  const size_t w1_base = (size_t)e * D_DIM * F_DIM;
  const size_t w2_base = (size_t)e * F_DIM * D_DIM;

  for (int fc = 0; fc < F_DIM; fc += FC) {
    __syncthreads();  // previous phase-B done before restaging
    // Stage W1 panel [k=0..767][n=fc..fc+63] transposed -> W1s[n][k].
    // float4 along n (coalesced), scattered b16 stores across 4 LDS rows.
    for (int i = tid * 4; i < FC * D_DIM; i += 256 * 4) {
      const int nn = i & (FC - 1), k = i >> 6;
      float4 v = *reinterpret_cast<const float4*>(
          &w1[w1_base + (size_t)k * F_DIM + fc + nn]);
      W1s[(size_t)(nn + 0) * W1S_ROW + k] = __float2bfloat16(v.x);
      W1s[(size_t)(nn + 1) * W1S_ROW + k] = __float2bfloat16(v.y);
      W1s[(size_t)(nn + 2) * W1S_ROW + k] = __float2bfloat16(v.z);
      W1s[(size_t)(nn + 3) * W1S_ROW + k] = __float2bfloat16(v.w);
    }
    // Stage W2 panel [k=fc..fc+63][n=0..767] transposed -> W2s[n][k].
    for (int i = tid * 4; i < FC * D_DIM; i += 256 * 4) {
      const int nn = i % D_DIM, kk = i / D_DIM;
      float4 v = *reinterpret_cast<const float4*>(
          &w2[w2_base + (size_t)(fc + kk) * D_DIM + nn]);
      W2s[(size_t)(nn + 0) * W2S_ROW + kk] = __float2bfloat16(v.x);
      W2s[(size_t)(nn + 1) * W2S_ROW + kk] = __float2bfloat16(v.y);
      W2s[(size_t)(nn + 2) * W2S_ROW + kk] = __float2bfloat16(v.z);
      W2s[(size_t)(nn + 3) * W2S_ROW + kk] = __float2bfloat16(v.w);
    }

    // Prefetch next chunk's weight panels into cache while WMMA runs.
    if (fc + FC < F_DIM) {
      const int fn = fc + FC;
      for (int k = tid; k < D_DIM; k += 256) {          // next W1 panel rows
        const float* p = &w1[w1_base + (size_t)k * F_DIM + fn];
        __builtin_prefetch(p, 0, 1);
        __builtin_prefetch(p + 32, 0, 1);               // 2nd 128B of 256B row
      }
      for (int i = tid; i < FC * (D_DIM / 32); i += 256) {  // next W2 panel
        const int kk = i / (D_DIM / 32), seg = i % (D_DIM / 32);
        __builtin_prefetch(&w2[w2_base + (size_t)(fn + kk) * D_DIM + seg * 32],
                           0, 1);
      }
    }
    __syncthreads();

    // Phase A: h1 chunk [32 x 64] = relu(Xs @ W1panel); 8 waves -> 8 tiles.
    {
      const int rtA = wv & 1;
      const int cf = wv >> 1;  // 0..3
      v8f h = (v8f)(0.f);
#pragma unroll 4
      for (int ks = 0; ks < D_DIM; ks += 32) {
        v16bf a = lds_load_a(Xs, D_DIM, rtA * 16, ks, lane);
        v16bf b = lds_load_b(W1s, W1S_ROW, cf * 16, ks, lane);
        h = __builtin_amdgcn_wmma_f32_16x16x32_bf16(false, a, false, b,
                                                    (short)0, h, false, false);
      }
      // ReLU + bf16 store to H1s (C layout: lane L -> N=L&15, M=j+8*(L>=16))
      const int nH = cf * 16 + (lane & 15);
      const int mhi = (lane >> 4) * 8;
#pragma unroll
      for (int j = 0; j < 8; ++j) {
        const int m = rtA * 16 + j + mhi;
        H1s[(size_t)m * FC + nH] = __float2bfloat16(fmaxf(h[j], 0.f));
      }
    }
    __syncthreads();

    // Phase B: acc += H1s @ W2panel  (K = 64 -> two 16x16x32 WMMAs per tile)
#pragma unroll
    for (int ks = 0; ks < FC; ks += 32) {
      v16bf a = lds_load_a(H1s, FC, rt * 16, ks, lane);
#pragma unroll
      for (int t = 0; t < 12; ++t) {
        v16bf b = lds_load_b(W2s, W2S_ROW, (cbase + t) * 16, ks, lane);
        acc[t] = __builtin_amdgcn_wmma_f32_16x16x32_bf16(false, a, false, b,
                                                         (short)0, acc[t],
                                                         false, false);
      }
    }
  }
  __syncthreads();

  // Epilogue: scale by router prob, scatter rows to out by token id.
  const int nlo = lane & 15;
  const int mhi = (lane >> 4) * 8;
#pragma unroll
  for (int t = 0; t < 12; ++t) {
    const int n = (cbase + t) * 16 + nlo;
#pragma unroll
    for (int j = 0; j < 8; ++j) {
      const int m = rt * 16 + j + mhi;
      if (m < rows)
        out[(size_t)tok_s[m] * D_DIM + n] = acc[t][j] * prob_s[m];
    }
  }
}

// ---- Host launch ----------------------------------------------------------
extern "C" void kernel_launch(void* const* d_in, const int* in_sizes, int n_in,
                              void* d_out, int out_size, void* d_ws, size_t ws_size,
                              hipStream_t stream) {
  const float* x        = (const float*)d_in[0];  // [B,S,D]
  const float* w_router = (const float*)d_in[1];  // [D,E]
  const float* w1       = (const float*)d_in[2];  // [E,D,F]
  const float* w2       = (const float*)d_in[3];  // [E,F,D]

  float* out = (float*)d_out;                          // [B,S,D]
  float* logits_out = out + (size_t)T_TOK * D_DIM;     // [B,S,E]
  int*   idx_out = (int*)(logits_out + (size_t)T_TOK * E_NUM);  // [B,S]

  float* prob_ws = (float*)d_ws;               // [T]
  int*   counts  = (int*)(prob_ws + T_TOK);    // [E]
  int*   bucket  = counts + E_NUM;             // [E][T]

  moe_zero_counts<<<1, 32, 0, stream>>>(counts);
  moe_router<<<T_TOK / 8, 256, 0, stream>>>(x, w_router, logits_out, idx_out,
                                            prob_ws, counts, bucket);
  dim3 grid(T_TOK / TILE_M, E_NUM);   // worst case: one expert owns all tokens
  moe_expert_ffn<<<grid, 256, 0, stream>>>(x, w1, w2, prob_ws, counts, bucket, out);
}
